// SDE_Net_72533407695045
// MI455X (gfx1250) — matrix operand order repaired and checked by reading, not compiled, not run on hardware
//
#include <hip/hip_runtime.h>
#include <hip/hip_bf16.h>
#include <math.h>

// ---------------------------------------------------------------------------
// MI455X (gfx1250) implementation. bf16 WMMA (v_wmma_f32_16x16x32_bf16) for
// all conv (via im2col) and linear layers; f32 accumulate; wave32.
// W tiles staged into LDS via the Tensor Data Mover (tensor_load_to_lds,
// TENSORcnt double-buffering) when the toolchain exposes it.
// ---------------------------------------------------------------------------

typedef __bf16 bf16;
typedef __attribute__((ext_vector_type(16))) __bf16 v16bf;
typedef __attribute__((ext_vector_type(8)))  __bf16 v8bf;
typedef __attribute__((ext_vector_type(8)))  float   v8f;
typedef unsigned int u32x4 __attribute__((ext_vector_type(4)));
typedef int          i32x4 __attribute__((ext_vector_type(4)));
typedef int          i32x8 __attribute__((ext_vector_type(8)));

#if defined(__HIP_DEVICE_COMPILE__) && \
    __has_builtin(__builtin_amdgcn_tensor_load_to_lds) && \
    __has_builtin(__builtin_amdgcn_s_wait_tensorcnt)
#define HAVE_TDM 1
#else
#define HAVE_TDM 0
#endif

#define BN_EPS 1e-5f

// ---------------- epilogue modes ----------------
#define EPI_BN_NCHW       0  // outB NCHW: bf16(scale[n]*acc+shift[n])
#define EPI_BN_RELU_NCHW  1  // outB NCHW: bf16(relu(scale[n]*acc+shift[n]))
#define EPI_BIAS_RELU_ROW 2  // outB row-major: bf16(relu(acc+bias[n]))
#define EPI_EULER_TANH    3  // yio[m*Npad+n] += 0.1f*tanhf(acc+bias[n])
#define EPI_BIAS_F32_ROW  4  // outF row-major ldc=Nreal, guard n<Nreal

// ---------------------------------------------------------------------------
// Elementwise / data-movement kernels
// ---------------------------------------------------------------------------
__global__ void k_f32_to_bf16(const float* __restrict__ in, bf16* __restrict__ out, long n) {
  long i = (long)blockIdx.x * blockDim.x + threadIdx.x;
  if (i < n) out[i] = (bf16)in[i];
}

// f32 [Nr x Kin] -> bf16 [Npad x Kpad] (zero padded)
__global__ void k_convert_pad_w(const float* __restrict__ in, bf16* __restrict__ out,
                                int Nr, int Kin, int Npad, int Kpad) {
  long i = (long)blockIdx.x * blockDim.x + threadIdx.x;
  long tot = (long)Npad * Kpad;
  if (i >= tot) return;
  int n = (int)(i / Kpad), k = (int)(i % Kpad);
  float v = (n < Nr && k < Kin) ? in[(long)n * Kin + k] : 0.f;
  out[i] = (bf16)v;
}

__global__ void k_bn_fold(const float* __restrict__ g, const float* __restrict__ b,
                          const float* __restrict__ m, const float* __restrict__ v,
                          float* __restrict__ scale, float* __restrict__ shift, int C) {
  int c = blockIdx.x * blockDim.x + threadIdx.x;
  if (c >= C) return;
  float s = g[c] * rsqrtf(v[c] + BN_EPS);
  scale[c] = s;
  shift[c] = b[c] - m[c] * s;
}

// bf16 NCHW -> bf16 im2col patch matrix [B*OH*OW x Kpad]
__global__ void k_im2col(const bf16* __restrict__ in, bf16* __restrict__ out,
                         int B, int Cin, int H, int W, int OH, int OW,
                         int KH, int KW, int stride, int pad, int Kpad) {
  long i = (long)blockIdx.x * blockDim.x + threadIdx.x;
  long M = (long)B * OH * OW;
  if (i >= M * Kpad) return;
  long mm = i / Kpad;
  int  k  = (int)(i % Kpad);
  int  K  = Cin * KH * KW;
  bf16 v = (bf16)0.f;
  if (k < K) {
    int b = (int)(mm / ((long)OH * OW));
    int r = (int)(mm % ((long)OH * OW));
    int p = r / OW, q = r % OW;
    int ci = k / (KH * KW);
    int rr = k % (KH * KW);
    int kh = rr / KW, kw = rr % KW;
    int ih = p * stride - pad + kh;
    int iw = q * stride - pad + kw;
    if (ih >= 0 && ih < H && iw >= 0 && iw < W)
      v = in[(((long)b * Cin + ci) * H + ih) * W + iw];
  }
  out[i] = v;
}

// 3x3 / stride2 / pad1 maxpool, bf16 NCHW, optional fused relu
__global__ void k_maxpool(const bf16* __restrict__ in, bf16* __restrict__ out,
                          int B, int C, int H, int W, int OH, int OW, int do_relu) {
  long i = (long)blockIdx.x * blockDim.x + threadIdx.x;
  long tot = (long)B * C * OH * OW;
  if (i >= tot) return;
  int q = (int)(i % OW); long t = i / OW;
  int p = (int)(t % OH); t /= OH;
  int c = (int)(t % C);
  int b = (int)(t / C);
  int h0 = p * 2 - 1, w0 = q * 2 - 1;
  float mx = -INFINITY;
  #pragma unroll
  for (int dh = 0; dh < 3; dh++)
    #pragma unroll
    for (int dw = 0; dw < 3; dw++) {
      int ih = h0 + dh, iw = w0 + dw;
      if (ih >= 0 && ih < H && iw >= 0 && iw < W)
        mx = fmaxf(mx, (float)in[(((long)b * C + c) * H + ih) * W + iw]);
    }
  if (do_relu) mx = fmaxf(mx, 0.f);
  out[i] = (bf16)mx;
}

// global average pool: bf16 [BC x HW] -> f32 [BC]
__global__ void k_avgpool(const bf16* __restrict__ in, float* __restrict__ y, int BC, int HW) {
  int i = blockIdx.x * blockDim.x + threadIdx.x;
  if (i >= BC) return;
  float s = 0.f;
  for (int j = 0; j < HW; j++) s += (float)in[(long)i * HW + j];
  y[i] = s / (float)HW;
}

// build drift input: [M x Kpad] bf16 = [sin t, cos t, y(0..D-1), 0...]
__global__ void k_ode_prepare(const float* __restrict__ y, bf16* __restrict__ Ain,
                              float t, int M, int D, int Kpad) {
  long i = (long)blockIdx.x * blockDim.x + threadIdx.x;
  if (i >= (long)M * Kpad) return;
  int m = (int)(i / Kpad), k = (int)(i % Kpad);
  float v;
  if (k == 0)            v = sinf(t);
  else if (k == 1)       v = cosf(t);
  else if (k < D + 2)    v = y[(long)m * D + (k - 2)];
  else                   v = 0.f;
  Ain[i] = (bf16)v;
}

// ---------------------------------------------------------------------------
// WMMA GEMM: C[M x Npad] = A[M x K] * W[Npad x K]^T
//   K multiple of 32, M multiple of 64, Npad multiple of 64.
//   Block = 128 threads (4 waves); block tile 64(M) x 64(N);
//   each wave: 16x64 (one A fragment -> 4 back-to-back WMMAs).
//   W slab (64 rows x 32 K = 4KB) staged in LDS, double buffered:
//   TDM tensor_load_to_lds + s_wait_tensorcnt when available, else
//   cooperative global_load + ds_store.
//
// Fragment layouts (ISA 7.12.2, wave32):
//  A 16x32 bf16: lane m=lane&15, half=lane>>4; elems 0..7 -> K=k0+half*8+{0..7},
//                elems 8..15 -> K=k0+16+half*8+{0..7}  (two 16B loads)
//  B 32x16 bf16 (from W[n][k]=B^T): lane n=lane&15, half=lane>>4;
//                elems 0..7 -> K=k0+half*8+{0..7}, 8..15 -> K=k0+16+half*8+{0..7}   (two 16B loads)
//  C/D f32:      row = r + 8*half, col = lane&15 for VGPR r.
// ---------------------------------------------------------------------------
__global__ __launch_bounds__(128)
void k_wmma_gemm_smem(const bf16* __restrict__ A, const bf16* __restrict__ Wt,
                      const float* __restrict__ aux0, const float* __restrict__ aux1,
                      float* __restrict__ outF, bf16* __restrict__ outB,
                      float* __restrict__ yio,
                      int M, int Npad, int K, int HW, int Nreal, int mode) {
  __shared__ __align__(16) bf16 sB[2][64 * 32];   // [buf][row*32 + k], 4KB per buffer

  const int lane = threadIdx.x & 31;
  const int wave = threadIdx.x >> 5;
  const int half = lane >> 4;
  const int l15  = lane & 15;

  const int nblk  = Npad >> 6;
  const int bm    = (int)(blockIdx.x / nblk) << 6;
  const int bn    = (int)(blockIdx.x % nblk) << 6;
  const int mbase = bm + wave * 16;

  const bf16* ap = A + (size_t)(mbase + l15) * K + half * 8;
  const int nchunks = K >> 5;

  // stage W slab for K-chunk c into sB[c&1]
  auto fill = [&](int c) {
    const int buf = c & 1;
    const int k0  = c << 5;
#if HAVE_TDM
    if (wave == 0) {
      // ---- Tensor DMA descriptor (D#), 2-D tile: 64 rows x 32 elems, bf16 ----
      unsigned long long ga = (unsigned long long)(size_t)(Wt + (size_t)bn * K + k0);
      // generic->LDS: hardware LDS byte address is addr[31:0] (flat aperture rule)
      unsigned lds = (unsigned)(size_t)&sB[buf][0];
      u32x4 g0;
      g0.x = 1u;                                             // count=1, user mode
      g0.y = lds;                                            // lds_addr
      g0.z = (unsigned)ga;                                   // global_addr[31:0]
      g0.w = (unsigned)((ga >> 32) & 0x01FFFFFFu)            // global_addr[56:32]
             | 0x80000000u;                                  // type=2 ("image")
      i32x8 g1;
      g1[0] = 0x00010000;            // workgroup_mask=0, data_size=1 (2 bytes)
      g1[1] = (int)0xFFFF0000u;      // tensor_dim0 = 0x7FFFFFFF (low16 in [31:16])
      g1[2] = (int)0xFFFF7FFFu;      // dim0 hi16 | tensor_dim1 low16 (dim1=0x7FFFFFFF)
      g1[3] = 0x00207FFF;            // dim1 hi16 | tile_dim0 = 32
      g1[4] = 64;                    // tile_dim1 = 64, tile_dim2 = 0
      g1[5] = K;                     // tensor_dim0_stride[31:0] (elements)
      g1[6] = 0;                     // stride hi | dim1_stride lo
      g1[7] = 0;
      i32x4 z4 = {0, 0, 0, 0};
#if defined(__clang_major__) && (__clang_major__ >= 23)
      i32x8 z8 = {0, 0, 0, 0, 0, 0, 0, 0};
      __builtin_amdgcn_tensor_load_to_lds(g0, g1, z4, z4, z8, 0);
#else
      __builtin_amdgcn_tensor_load_to_lds(g0, g1, z4, z4, 0);
#endif
    }
#else
    // cooperative fallback: 128 threads x 32B = 4KB
    const int t = threadIdx.x;
    #pragma unroll
    for (int p = t * 2; p < t * 2 + 2; p++) {
      const int row = p >> 2, sub = p & 3;
      const bf16* g = Wt + (size_t)(bn + row) * K + k0 + sub * 8;
      *(v8bf*)&sB[buf][row * 32 + sub * 8] = *(const v8bf*)g;
    }
#endif
  };

  v8f acc[4] = {v8f{}, v8f{}, v8f{}, v8f{}};

  fill(0);
  for (int c = 0; c < nchunks; c++) {
    const int buf = c & 1;
    const int k0  = c << 5;
    __syncthreads();                       // everyone done reading buf^1
    if (c + 1 < nchunks) {
      fill(c + 1);
      __builtin_prefetch(ap + k0 + 32, 0, 0);   // next A slab -> global_prefetch
    }
#if HAVE_TDM
    if (wave == 0) {
      if (c + 1 < nchunks) __builtin_amdgcn_s_wait_tensorcnt(1);  // chunk c arrived
      else                 __builtin_amdgcn_s_wait_tensorcnt(0);
    }
#endif
    __syncthreads();                       // chunk c visible to all waves

    v8bf a0 = *(const v8bf*)(ap + k0);
    v8bf a1 = *(const v8bf*)(ap + k0 + 16);
    v16bf afrag = __builtin_shufflevector(a0, a1, 0,1,2,3,4,5,6,7,8,9,10,11,12,13,14,15);

    #pragma unroll
    for (int g = 0; g < 4; g++) {
      const bf16* bp = &sB[buf][(g * 16 + l15) * 32 + half * 8];
      v8bf b0 = *(const v8bf*)bp;          // ds_load_b128
      v8bf b1 = *(const v8bf*)(bp + 16);   // ds_load_b128
      v16bf bfrag = __builtin_shufflevector(b0, b1, 0,1,2,3,4,5,6,7,8,9,10,11,12,13,14,15);
      acc[g] = __builtin_amdgcn_wmma_f32_16x16x32_bf16(
          false, afrag, false, bfrag, (short)0, acc[g], false, false);
    }
  }

  // ---------------- epilogue ----------------
  #pragma unroll
  for (int g = 0; g < 4; g++) {
    const int n = bn + g * 16 + l15;
    #pragma unroll
    for (int r = 0; r < 8; r++) {
      const int m = mbase + r + half * 8;
      float v = acc[g][r];
      if (mode == EPI_BN_NCHW || mode == EPI_BN_RELU_NCHW) {
        float sv = aux0[n] * v + aux1[n];
        if (mode == EPI_BN_RELU_NCHW) sv = fmaxf(sv, 0.f);
        long o = ((long)(m / HW) * Npad + n) * HW + (m % HW);   // (b,p,q),co -> NCHW
        outB[o] = (bf16)sv;
      } else if (mode == EPI_BIAS_RELU_ROW) {
        outB[(long)m * Npad + n] = (bf16)fmaxf(v + aux0[n], 0.f);
      } else if (mode == EPI_EULER_TANH) {
        yio[(long)m * Npad + n] += 0.1f * tanhf(v + aux0[n]);
      } else { // EPI_BIAS_F32_ROW
        if (n < Nreal) outF[(long)m * Nreal + n] = v + aux0[n];
      }
    }
  }
}

// ---------------------------------------------------------------------------
// Host side
// ---------------------------------------------------------------------------
static inline unsigned grid1(long n, int b) { return (unsigned)((n + b - 1) / b); }

static inline void launch_gemm(const bf16* A, const bf16* Wt,
                               const float* aux0, const float* aux1,
                               float* outF, bf16* outB, float* yio,
                               int M, int Npad, int K, int HW, int Nreal, int mode,
                               hipStream_t st) {
  unsigned blocks = (unsigned)((long)(M / 64) * (Npad / 64));
  k_wmma_gemm_smem<<<blocks, 128, 0, st>>>(A, Wt, aux0, aux1, outF, outB, yio,
                                           M, Npad, K, HW, Nreal, mode);
}

extern "C" void kernel_launch(void* const* d_in, const int* in_sizes, int n_in,
                              void* d_out, int out_size, void* d_ws, size_t ws_size,
                              hipStream_t stream) {
  (void)in_sizes; (void)n_in; (void)out_size; (void)ws_size;

  // ---- inputs (depth-first insertion order of setup_inputs dict) ----
  const float* x        = (const float*)d_in[0];   // [64,3,224,224]
  const float* conv1_w  = (const float*)d_in[1];   // [64,3,7,7]
  const float* bn1_g    = (const float*)d_in[2];
  const float* bn1_b    = (const float*)d_in[3];
  const float* bn1_m    = (const float*)d_in[4];
  const float* bn1_v    = (const float*)d_in[5];
  const float* blk_w[4];
  const float* blk_bn[4][4];
  for (int i = 0; i < 4; i++) {
    blk_w[i] = (const float*)d_in[6 + 5 * i];
    for (int j = 0; j < 4; j++) blk_bn[i][j] = (const float*)d_in[7 + 5 * i + j];
  }
  const float* lin_in_w  = (const float*)d_in[26]; // [1024,1026]
  const float* lin_in_b  = (const float*)d_in[27];
  const float* hid_w     = (const float*)d_in[28]; // [1024,1024]
  const float* hid_b     = (const float*)d_in[29];
  const float* lin_out_w = (const float*)d_in[30]; // [1024,1024]
  const float* lin_out_b = (const float*)d_in[31];
  const float* fc_w      = (const float*)d_in[32]; // [1000,1024]
  const float* fc_b      = (const float*)d_in[33];
  float* out = (float*)d_out;                      // [64,1000]

  // ---- workspace carve-up ----
  char* ws = (char*)d_ws;
  size_t off = 0;
  auto alloc = [&](size_t bytes) -> void* {
    void* p = ws + off;
    off += (bytes + 255) & ~(size_t)255;
    return p;
  };

  const int B = 64;
  bf16*  x_bf   = (bf16*)alloc((size_t)B * 3 * 224 * 224 * 2);
  bf16*  colbuf = (bf16*)alloc((size_t)802816 * 160 * 2);       // max im2col (stem)
  bf16*  actA   = (bf16*)alloc((size_t)B * 64 * 112 * 112 * 2); // max conv out
  bf16*  actB   = (bf16*)alloc((size_t)B * 64 * 56 * 56 * 2);   // max pooled out
  // conv weights (bf16, K padded)
  const int cw_N[5]  = {64, 128, 256, 512, 1024};
  const int cw_K[5]  = {147, 576, 1152, 2304, 4608};
  const int cw_Kp[5] = {160, 576, 1152, 2304, 4608};
  bf16* cwb[5];
  for (int i = 0; i < 5; i++) cwb[i] = (bf16*)alloc((size_t)cw_N[i] * cw_Kp[i] * 2);
  float *bns[5], *bnh[5];
  for (int i = 0; i < 5; i++) {
    bns[i] = (float*)alloc((size_t)cw_N[i] * 4);
    bnh[i] = (float*)alloc((size_t)cw_N[i] * 4);
  }
  bf16* Win_bf  = (bf16*)alloc((size_t)1024 * 1056 * 2);
  bf16* Wh_bf   = (bf16*)alloc((size_t)1024 * 1024 * 2);
  bf16* Wout_bf = (bf16*)alloc((size_t)1024 * 1024 * 2);
  bf16* Wfc_bf  = (bf16*)alloc((size_t)1024 * 1024 * 2);        // padded to 1024 rows
  float* y      = (float*)alloc((size_t)B * 1024 * 4);
  bf16*  Ain    = (bf16*)alloc((size_t)B * 1056 * 2);
  bf16*  h1     = (bf16*)alloc((size_t)B * 1024 * 2);
  bf16*  h2     = (bf16*)alloc((size_t)B * 1024 * 2);
  bf16*  ybf    = (bf16*)alloc((size_t)B * 1024 * 2);

  // ---- per-call weight conversions ----
  {
    long n = (long)B * 3 * 224 * 224;
    k_f32_to_bf16<<<grid1(n, 256), 256, 0, stream>>>(x, x_bf, n);
  }
  k_convert_pad_w<<<grid1((long)64 * 160, 256), 256, 0, stream>>>(conv1_w, cwb[0], 64, 147, 64, 160);
  for (int i = 0; i < 4; i++) {
    long tot = (long)cw_N[i + 1] * cw_Kp[i + 1];
    k_convert_pad_w<<<grid1(tot, 256), 256, 0, stream>>>(blk_w[i], cwb[i + 1],
                                                         cw_N[i + 1], cw_K[i + 1],
                                                         cw_N[i + 1], cw_Kp[i + 1]);
  }
  k_bn_fold<<<grid1(64, 64), 64, 0, stream>>>(bn1_g, bn1_b, bn1_m, bn1_v, bns[0], bnh[0], 64);
  for (int i = 0; i < 4; i++) {
    int C = cw_N[i + 1];
    k_bn_fold<<<grid1(C, 64), 64, 0, stream>>>(blk_bn[i][0], blk_bn[i][1],
                                               blk_bn[i][2], blk_bn[i][3],
                                               bns[i + 1], bnh[i + 1], C);
  }
  k_convert_pad_w<<<grid1((long)1024 * 1056, 256), 256, 0, stream>>>(lin_in_w, Win_bf, 1024, 1026, 1024, 1056);
  k_convert_pad_w<<<grid1((long)1024 * 1024, 256), 256, 0, stream>>>(hid_w, Wh_bf, 1024, 1024, 1024, 1024);
  k_convert_pad_w<<<grid1((long)1024 * 1024, 256), 256, 0, stream>>>(lin_out_w, Wout_bf, 1024, 1024, 1024, 1024);
  k_convert_pad_w<<<grid1((long)1024 * 1024, 256), 256, 0, stream>>>(fc_w, Wfc_bf, 1000, 1024, 1024, 1024);

  // ---- conv stages ----
  const int st_Cin[5]  = {3, 64, 128, 256, 512};
  const int st_H[5]    = {224, 56, 28, 14, 7};
  const int st_OH[5]   = {112, 56, 28, 14, 7};
  const int st_KH[5]   = {7, 3, 3, 3, 3};
  const int st_str[5]  = {2, 1, 1, 1, 1};
  const int st_pad[5]  = {3, 1, 1, 1, 1};
  const int st_POH[5]  = {56, 28, 14, 7, 4};

  for (int s = 0; s < 5; s++) {
    const bf16* src = (s == 0) ? x_bf : actB;
    int Cin = st_Cin[s], H = st_H[s], OH = st_OH[s];
    int Cout = cw_N[s], Kp = cw_Kp[s];
    long M = (long)B * OH * OH;
    k_im2col<<<grid1(M * Kp, 256), 256, 0, stream>>>(src, colbuf, B, Cin, H, H, OH, OH,
                                                     st_KH[s], st_KH[s], st_str[s], st_pad[s], Kp);
    int mode = (s == 0) ? EPI_BN_RELU_NCHW : EPI_BN_NCHW;
    launch_gemm(colbuf, cwb[s], bns[s], bnh[s], nullptr, actA, nullptr,
                (int)M, Cout, Kp, OH * OH, Cout, mode, stream);
    int POH = st_POH[s];
    long ptot = (long)B * Cout * POH * POH;
    k_maxpool<<<grid1(ptot, 256), 256, 0, stream>>>(actA, actB, B, Cout, OH, OH, POH, POH,
                                                    (s == 0) ? 0 : 1);
  }

  // ---- global average pool -> y ----
  k_avgpool<<<grid1((long)B * 1024, 256), 256, 0, stream>>>(actB, y, B * 1024, 16);

  // ---- 90 Euler steps of the drift MLP ----
  for (int s = 0; s < 90; s++) {
    float t = 0.1f * (float)s;
    k_ode_prepare<<<grid1((long)B * 1056, 256), 256, 0, stream>>>(y, Ain, t, B, 1024, 1056);
    launch_gemm(Ain, Win_bf, lin_in_b, nullptr, nullptr, h1, nullptr,
                B, 1024, 1056, 0, 1024, EPI_BIAS_RELU_ROW, stream);
    launch_gemm(h1, Wh_bf, hid_b, nullptr, nullptr, h2, nullptr,
                B, 1024, 1024, 0, 1024, EPI_BIAS_RELU_ROW, stream);
    launch_gemm(h2, Wout_bf, lin_out_b, nullptr, nullptr, nullptr, y,
                B, 1024, 1024, 0, 1024, EPI_EULER_TANH, stream);
  }

  // ---- final FC ----
  {
    long n = (long)B * 1024;
    k_f32_to_bf16<<<grid1(n, 256), 256, 0, stream>>>(y, ybf, n);
    launch_gemm(ybf, Wfc_bf, fc_b, nullptr, out, nullptr, nullptr,
                B, 1024, 1024, 0, 1000, EPI_BIAS_F32_ROW, stream);
  }
}
